// Neg_Attn_73332271612517
// MI455X (gfx1250) — matrix operand order repaired and checked by reading, not compiled, not run on hardware
//
#include <hip/hip_runtime.h>
#include <hip/hip_bf16.h>

// Problem dims (fixed by reference)
#define B_  2
#define H_  4
#define BH_ 8
#define N_  1024
#define M_  4096
#define D_  64
#define C_  256

// GEMM tiling
#define BM 32
#define BN 64
#define BK 32
#define LDA 34   // f32 LDS row stride (floats): 136B keeps 8B alignment for b64 frags
#define SA  40   // bf16 LDS row stride (elems): 80B keeps 16B alignment for b128 frags

typedef float v2f __attribute__((ext_vector_type(2)));
typedef float v8f __attribute__((ext_vector_type(8)));
typedef __bf16 bf16;
typedef __bf16 v4bf  __attribute__((ext_vector_type(4)));
typedef __bf16 v8bf  __attribute__((ext_vector_type(8)));
typedef __bf16 v16bf __attribute__((ext_vector_type(16)));

// ---------------------------------------------------------------------------
// f32 WMMA chunk (16x16 tile, K=32): 8x v_wmma_f32_16x16x4_f32.
__device__ __forceinline__ v8f wmma_chunk_f32(const float* __restrict__ Arow,
                                              const float* __restrict__ Brow,
                                              int hi, v8f acc) {
#pragma unroll
  for (int ks = 0; ks < BK; ks += 4) {
    v2f a  = *(const v2f*)(Arow + ks + 2 * hi);
    v2f bb = *(const v2f*)(Brow + ks + 2 * hi);
    acc = __builtin_amdgcn_wmma_f32_16x16x4_f32(false, a, false, bb,
                                                (short)0, acc, false, false);
  }
  return acc;
}

// bf16 fragment assembly: two 16B LDS runs -> one 16-elem fragment (8 VGPRs)
__device__ __forceinline__ v16bf ld16(const bf16* p0, const bf16* p1) {
  v8bf x = *(const v8bf*)p0;
  v8bf y = *(const v8bf*)p1;
  return __builtin_shufflevector(x, y, 0, 1, 2, 3, 4, 5, 6, 7,
                                 8, 9, 10, 11, 12, 13, 14, 15);
}

// bf16x3 fp32-emulated chunk (16x16 tile, K=32): 3x v_wmma_f32_16x16x32_bf16.
// A and B lane-half hi cover K runs {8hi..8hi+7} and {16+8hi..16+8hi+7} (ISA 16-bit
// 16x32 layout).
__device__ __forceinline__ v8f wmma_chunk_bf16(const bf16* __restrict__ Ah,
                                               const bf16* __restrict__ Al,
                                               const bf16* __restrict__ Bh,
                                               const bf16* __restrict__ Bl,
                                               int hi, v8f acc) {
  v16bf ah = ld16(Ah + 8 * hi, Ah + 16 + 8 * hi);
  v16bf al = ld16(Al + 8 * hi, Al + 16 + 8 * hi);
  v16bf bh = ld16(Bh + 8 * hi, Bh + 16 + 8 * hi);
  v16bf bl = ld16(Bl + 8 * hi, Bl + 16 + 8 * hi);
  acc = __builtin_amdgcn_wmma_f32_16x16x32_bf16(false, ah, false, bh,
                                                (short)0, acc, false, false);
  acc = __builtin_amdgcn_wmma_f32_16x16x32_bf16(false, ah, false, bl,
                                                (short)0, acc, false, false);
  acc = __builtin_amdgcn_wmma_f32_16x16x32_bf16(false, al, false, bh,
                                                (short)0, acc, false, false);
  return acc;
}

__device__ __forceinline__ void split_bf(float f, bf16& h, bf16& l) {
  h = (bf16)f;                 // RNE truncation (v_cvt)
  l = (bf16)(f - (float)h);    // residual
}

// ---------------------------------------------------------------------------
// Kernel 0a: zero the halo columns of Xpad (BH, C, N+2)
__global__ __launch_bounds__(256) void k_zero_pad(float* __restrict__ Xpad) {
  int i = blockIdx.x * 256 + threadIdx.x;   // over BH*C rows
  if (i < BH_ * C_) {
    Xpad[(size_t)i * (N_ + 2)]          = 0.f;
    Xpad[(size_t)i * (N_ + 2) + N_ + 1] = 0.f;
  }
}

// Kernel 0b: Wc_out (M,C,3) -> A2 (M, 3C) with kk = t*C + c
__global__ __launch_bounds__(256) void k_wcout(const float* __restrict__ Wc_out,
                                               float* __restrict__ A2) {
  int i = blockIdx.x * 256 + threadIdx.x;   // over M*3C
  int m = i / (3 * C_);
  int rem = i - m * (3 * C_);
  int t = rem / C_;
  int c = rem - t * C_;
  A2[i] = Wc_out[((size_t)m * C_ + c) * 3 + t];
}

// ---------------------------------------------------------------------------
// Kernel 1 (bf16x3): x = sigmoid(-y)*(Wc_in @ S_attn + bc_in) -> Xpad[:,:,1:N+1]
__global__ __launch_bounds__(256)
void k1_gemm_gate(const float* __restrict__ Wc_in, const float* __restrict__ S_attn,
                  const float* __restrict__ y, const float* __restrict__ bc_in,
                  float* __restrict__ Xpad) {
  __shared__ __attribute__((aligned(16))) bf16 Ash[BM * SA], Asl[BM * SA];
  __shared__ __attribute__((aligned(16))) bf16 Bsh[BN * SA], Bsl[BN * SA];
  const int b = blockIdx.z;
  const int rowBase = blockIdx.y * BM;          // over C
  const int colBase = blockIdx.x * BN;          // over N
  const int tid = threadIdx.x;
  const int w = tid >> 5, lane = tid & 31, hi = lane >> 4, ln = lane & 15;
  const int tr = (w >> 2) * 16, tc = (w & 3) * 16;
  const float* Bg = S_attn + (size_t)b * M_ * N_;
  const int ar = tid >> 3, ak = (tid & 7) * 4;
  v8f acc = {0.f, 0.f, 0.f, 0.f, 0.f, 0.f, 0.f, 0.f};

  for (int kBase = 0; kBase < M_; kBase += BK) {
    const float* Ag = Wc_in + (size_t)(rowBase + ar) * M_ + kBase + ak;
    v4bf hv, lv;
#pragma unroll
    for (int u = 0; u < 4; ++u) { bf16 h, l; split_bf(Ag[u], h, l); hv[u] = h; lv[u] = l; }
    *(v4bf*)&Ash[ar * SA + ak] = hv;
    *(v4bf*)&Asl[ar * SA + ak] = lv;
#pragma unroll
    for (int i = 0; i < 8; ++i) {
      int idx = tid + i * 256;
      int nl = idx & 63, kl = idx >> 6;
      bf16 h, l;
      split_bf(Bg[(size_t)(kBase + kl) * N_ + colBase + nl], h, l);
      Bsh[nl * SA + kl] = h;
      Bsl[nl * SA + kl] = l;
    }
    __syncthreads();
    acc = wmma_chunk_bf16(&Ash[(tr + ln) * SA], &Asl[(tr + ln) * SA],
                          &Bsh[(tc + ln) * SA], &Bsl[(tc + ln) * SA], hi, acc);
    __syncthreads();
  }
#pragma unroll
  for (int r = 0; r < 8; ++r) {
    int c = rowBase + tr + r + 8 * hi;
    int n = colBase + tc + ln;
    float g = 1.f / (1.f + __expf(y[((size_t)b * C_ + c) * N_ + n])); // sigmoid(-y)
    Xpad[((size_t)b * C_ + c) * (N_ + 2) + n + 1] = g * (acc[r] + bc_in[c]);
  }
}

// ---------------------------------------------------------------------------
// Kernel 2 (bf16x3): s1 = conv3(x) + bc_out as GEMM  A2(M x 3C) @ shifted Xpad
__global__ __launch_bounds__(256)
void k2_gemm_conv(const float* __restrict__ A2, const float* __restrict__ Xpad,
                  const float* __restrict__ bc_out, float* __restrict__ s1) {
  __shared__ __attribute__((aligned(16))) bf16 Ash[BM * SA], Asl[BM * SA];
  __shared__ __attribute__((aligned(16))) bf16 Bsh[BN * SA], Bsl[BN * SA];
  const int b = blockIdx.z;
  const int rowBase = blockIdx.y * BM;          // over M
  const int colBase = blockIdx.x * BN;          // over N
  const int tid = threadIdx.x;
  const int w = tid >> 5, lane = tid & 31, hi = lane >> 4, ln = lane & 15;
  const int tr = (w >> 2) * 16, tc = (w & 3) * 16;
  const int ar = tid >> 3, ak = (tid & 7) * 4;
  v8f acc = {0.f, 0.f, 0.f, 0.f, 0.f, 0.f, 0.f, 0.f};

  for (int kBase = 0; kBase < 3 * C_; kBase += BK) {
    const int t = kBase / C_;                   // BK | C: tap constant per chunk
    const int c0 = kBase - t * C_;
    const float* Ag = A2 + (size_t)(rowBase + ar) * (3 * C_) + kBase + ak;
    v4bf hv, lv;
#pragma unroll
    for (int u = 0; u < 4; ++u) { bf16 h, l; split_bf(Ag[u], h, l); hv[u] = h; lv[u] = l; }
    *(v4bf*)&Ash[ar * SA + ak] = hv;
    *(v4bf*)&Asl[ar * SA + ak] = lv;
#pragma unroll
    for (int i = 0; i < 8; ++i) {
      int idx = tid + i * 256;
      int nl = idx & 63, kl = idx >> 6;
      bf16 h, l;   // B'[t*C+c][n] = xpad[c][n+t]  (conv pad folded into halo)
      split_bf(Xpad[((size_t)b * C_ + c0 + kl) * (N_ + 2) + colBase + nl + t], h, l);
      Bsh[nl * SA + kl] = h;
      Bsl[nl * SA + kl] = l;
    }
    __syncthreads();
    acc = wmma_chunk_bf16(&Ash[(tr + ln) * SA], &Asl[(tr + ln) * SA],
                          &Bsh[(tc + ln) * SA], &Bsl[(tc + ln) * SA], hi, acc);
    __syncthreads();
  }
#pragma unroll
  for (int r = 0; r < 8; ++r) {
    int m = rowBase + tr + r + 8 * hi;
    int n = colBase + tc + ln;
    s1[((size_t)b * M_ + m) * N_ + n] = acc[r] + bc_out[m];
  }
}

// ---------------------------------------------------------------------------
// Kernel 3a: per-row softmax stats (wave32 shuffle reduction, 1 wave / row)
__global__ __launch_bounds__(256)
void k3a_stats(const float* __restrict__ s1, float* __restrict__ rowmax,
               float* __restrict__ rowinv) {
  const int w = threadIdx.x >> 5, lane = threadIdx.x & 31;
  const int row = blockIdx.x * 8 + w;           // over BH*M
  const float* p = s1 + (size_t)row * N_;
  float mx = -3.402823466e38f;
  for (int i = lane; i < N_; i += 32) mx = fmaxf(mx, p[i]);
#pragma unroll
  for (int off = 16; off; off >>= 1) mx = fmaxf(mx, __shfl_xor(mx, off, 32));
  float s = 0.f;
  for (int i = lane; i < N_; i += 32) s += __expf(p[i] - mx);
#pragma unroll
  for (int off = 16; off; off >>= 1) s += __shfl_xor(s, off, 32);
  if (lane == 0) { rowmax[row] = mx; rowinv[row] = 1.f / s; }
}

// Kernel 3b (f32 WMMA): hs = softmax(s1) @ S_value; exp fused into A-staging,
// B staged via CDNA5 async global->LDS copies (ASYNCcnt).
__global__ __launch_bounds__(256)
void k3b_gemm_attn(const float* __restrict__ s1, const float* __restrict__ S_value,
                   const float* __restrict__ rowmax, const float* __restrict__ rowinv,
                   float* __restrict__ hs) {
  __shared__ __attribute__((aligned(16))) float As[BM * LDA];
  __shared__ __attribute__((aligned(16))) float Bs[BN * LDA];
  const int b = blockIdx.z;
  const int rowBase = blockIdx.y * BM;          // over M
  const int tid = threadIdx.x;
  const int w = tid >> 5, lane = tid & 31, hi = lane >> 4, ln = lane & 15;
  const int tr = (w >> 2) * 16, tc = (w & 3) * 16;
  const int ar = tid >> 3, ak = (tid & 7) * 4;
  const float rmx = rowmax[(size_t)b * M_ + rowBase + ar];
  const float* Bg = S_value + (size_t)b * N_ * D_;
  v8f acc = {0.f, 0.f, 0.f, 0.f, 0.f, 0.f, 0.f, 0.f};

  for (int kBase = 0; kBase < N_; kBase += BK) {
#pragma unroll
    for (int u = 0; u < 4; ++u)
      As[ar * LDA + ak + u] =
          __expf(s1[((size_t)b * M_ + rowBase + ar) * N_ + kBase + ak + u] - rmx);
    // Async transposed staging: one f32 per lane straight into LDS, no VGPR hop.
#pragma unroll
    for (int i = 0; i < 8; ++i) {
      int idx = tid + i * 256;
      int nl = idx & 63, kl = idx >> 6;
      unsigned lds_addr = (unsigned)(size_t)&Bs[nl * LDA + kl];
      const float* gp = Bg + (size_t)(kBase + kl) * D_ + nl;
      asm volatile("global_load_async_to_lds_b32 %0, %1, off"
                   :: "v"(lds_addr), "v"(gp) : "memory");
    }
    asm volatile("s_wait_asynccnt 0" ::: "memory");
    __syncthreads();
    acc = wmma_chunk_f32(As + (tr + ln) * LDA, Bs + (tc + ln) * LDA, hi, acc);
    __syncthreads();
  }
#pragma unroll
  for (int r = 0; r < 8; ++r) {
    int m = rowBase + tr + r + 8 * hi;
    int d = tc + ln;
    hs[((size_t)b * M_ + m) * D_ + d] = acc[r] * rowinv[(size_t)b * M_ + m];
  }
}

// ---------------------------------------------------------------------------
// Kernel 4 (bf16x3): out[b, k, h*64+d] = W_out(k,:) @ hs[bh](:,d) + b_out[k]
__global__ __launch_bounds__(256)
void k4_gemm_out(const float* __restrict__ W_out, const float* __restrict__ hs,
                 const float* __restrict__ b_out, float* __restrict__ out) {
  __shared__ __attribute__((aligned(16))) bf16 Ash[BM * SA], Asl[BM * SA];
  __shared__ __attribute__((aligned(16))) bf16 Bsh[BN * SA], Bsl[BN * SA];
  const int bh = blockIdx.z;
  const int batch = bh >> 2, h = bh & 3;        // H_ == 4
  const int rowBase = blockIdx.y * BM;          // over M (k)
  const int tid = threadIdx.x;
  const int w = tid >> 5, lane = tid & 31, hi = lane >> 4, ln = lane & 15;
  const int tr = (w >> 2) * 16, tc = (w & 3) * 16;
  const int ar = tid >> 3, ak = (tid & 7) * 4;
  const float* Bg = hs + (size_t)bh * M_ * D_;
  v8f acc = {0.f, 0.f, 0.f, 0.f, 0.f, 0.f, 0.f, 0.f};

  for (int kBase = 0; kBase < M_; kBase += BK) {
    const float* Ag = W_out + (size_t)(rowBase + ar) * M_ + kBase + ak;
    v4bf hv, lv;
#pragma unroll
    for (int u = 0; u < 4; ++u) { bf16 hh, ll; split_bf(Ag[u], hh, ll); hv[u] = hh; lv[u] = ll; }
    *(v4bf*)&Ash[ar * SA + ak] = hv;
    *(v4bf*)&Asl[ar * SA + ak] = lv;
#pragma unroll
    for (int i = 0; i < 8; ++i) {
      int idx = tid + i * 256;
      int nl = idx & 63, kl = idx >> 6;
      bf16 hh, ll;
      split_bf(Bg[(size_t)(kBase + kl) * D_ + nl], hh, ll);
      Bsh[nl * SA + kl] = hh;
      Bsl[nl * SA + kl] = ll;
    }
    __syncthreads();
    acc = wmma_chunk_bf16(&Ash[(tr + ln) * SA], &Asl[(tr + ln) * SA],
                          &Bsh[(tc + ln) * SA], &Bsl[(tc + ln) * SA], hi, acc);
    __syncthreads();
  }
#pragma unroll
  for (int r = 0; r < 8; ++r) {
    int k = rowBase + tr + r + 8 * hi;
    int d = tc + ln;
    out[((size_t)batch * M_ + k) * (H_ * D_) + h * D_ + d] = acc[r] + b_out[k];
  }
}

// ---------------------------------------------------------------------------
extern "C" void kernel_launch(void* const* d_in, const int* in_sizes, int n_in,
                              void* d_out, int out_size, void* d_ws, size_t ws_size,
                              hipStream_t stream) {
  const float* S_value = (const float*)d_in[0];
  const float* S_attn  = (const float*)d_in[1];
  const float* y       = (const float*)d_in[2];
  const float* Wc_in   = (const float*)d_in[3];
  const float* bc_in   = (const float*)d_in[4];
  const float* Wc_out  = (const float*)d_in[5];
  const float* bc_out  = (const float*)d_in[6];
  const float* W_out   = (const float*)d_in[7];
  const float* b_out   = (const float*)d_in[8];
  // d_in[9] = h (==4), baked into H_

  float* out = (float*)d_out;                            // (B, M, H*D)
  float* s1  = out + (size_t)B_ * M_ * (H_ * D_);        // (BH, M, N)

  float* ws     = (float*)d_ws;
  float* Xpad   = ws;                                    // BH*C*(N+2)
  float* A2     = Xpad + (size_t)BH_ * C_ * (N_ + 2);    // M*3C
  float* hsbuf  = A2 + (size_t)M_ * 3 * C_;              // BH*M*D
  float* rowmax = hsbuf + (size_t)BH_ * M_ * D_;         // BH*M
  float* rowinv = rowmax + (size_t)BH_ * M_;             // BH*M

  k_zero_pad<<<(BH_ * C_ + 255) / 256, 256, 0, stream>>>(Xpad);
  k_wcout<<<(M_ * 3 * C_) / 256, 256, 0, stream>>>(Wc_out, A2);
  k1_gemm_gate<<<dim3(N_ / BN, C_ / BM, BH_), 256, 0, stream>>>(Wc_in, S_attn, y,
                                                                bc_in, Xpad);
  k2_gemm_conv<<<dim3(N_ / BN, M_ / BM, BH_), 256, 0, stream>>>(A2, Xpad, bc_out, s1);
  k3a_stats<<<(BH_ * M_) / 8, 256, 0, stream>>>(s1, rowmax, rowinv);
  k3b_gemm_attn<<<dim3(1, M_ / BM, BH_), 256, 0, stream>>>(s1, S_value, rowmax,
                                                           rowinv, hsbuf);
  k4_gemm_out<<<dim3(1, M_ / BM, BH_), 256, 0, stream>>>(W_out, hsbuf, b_out, out);
}
